// MambaBlock_55903294324790
// MI455X (gfx1250) — compile-verified, matrix operand order checked by reading, not run on hardware
//
#include <hip/hip_runtime.h>
#include <hip/hip_bf16.h>

#define SEQ     2048
#define DMODEL  1024
#define DINNER  2048
#define DSTATE  16
#define DTRANK  64
#define NPROJ   96      // DTRANK + 2*DSTATE
#define DIN2    4096    // 2*DINNER

typedef __attribute__((ext_vector_type(16))) _Float16 v16h;
typedef __attribute__((ext_vector_type(8)))  _Float16 v8h;
typedef __attribute__((ext_vector_type(8)))  float    v8f;
typedef int v4i __attribute__((vector_size(16)));   // matches builtin param type

#if __has_builtin(__builtin_amdgcn_global_load_async_to_lds_b128)
#define USE_ASYNC_LDS 1
#else
#define USE_ASYNC_LDS 0
#endif

// ---------------------------------------------------------------------------
// WMMA fragment helpers — layout per CDNA5 ISA (CDNA5 ISA, wave32):
// 16-bit A 16x32: lane m (0..15) holds row m, K = {k0..k0+7, k0+16..k0+23};
//                 lane m+16 holds row m, K = {k0+8..k0+15, k0+24..k0+31}.
// B (32x16) is column-major symmetric: lane n holds column n along K.
// ---------------------------------------------------------------------------
__device__ __forceinline__ v16h load_frag_k(const _Float16* __restrict__ p,
                                            int ld, int r0, int k0) {
  const int lane = threadIdx.x & 31;
  const int m    = lane & 15;
  const int half = lane >> 4;
  const _Float16* base = p + (size_t)(r0 + m) * ld + k0 + half * 8;
  v8h lo = *(const v8h*)(base);
  v8h hi = *(const v8h*)(base + 16);
  v16h v;
#pragma unroll
  for (int i = 0; i < 8; ++i) { v[i] = lo[i]; v[i + 8] = hi[i]; }
  return v;
}

// Fragment read from an LDS tile with fixed leading dimension 32.
__device__ __forceinline__ v16h load_frag_lds(const _Float16* p, int r0) {
  const int lane = threadIdx.x & 31;
  const int m    = lane & 15;
  const int half = lane >> 4;
  const _Float16* base = p + (r0 + m) * 32 + half * 8;
  v8h lo = *(const v8h*)(base);          // ds_load_b128
  v8h hi = *(const v8h*)(base + 16);     // ds_load_b128
  v16h v;
#pragma unroll
  for (int i = 0; i < 8; ++i) { v[i] = lo[i]; v[i + 8] = hi[i]; }
  return v;
}

// C/D 16x16 f32: VGPR r <-> M = r + 8*(lane>=16), N = lane&15.
// mode: 0 = plain store, 1 = +bias then softplus, 2 = +residual
__device__ __forceinline__ void store_acc(float* __restrict__ C, int ldc,
                                          int r0, int c0, v8f acc, int mode,
                                          const float* __restrict__ bias,
                                          const float* __restrict__ res, int ldres) {
  const int lane = threadIdx.x & 31;
  const int col  = c0 + (lane & 15);
  const int rb   = r0 + (lane >> 4) * 8;
#pragma unroll
  for (int r = 0; r < 8; ++r) {
    float v = acc[r];
    const int row = rb + r;
    if (mode == 1) {
      v += bias[col];
      v = (v > 20.f) ? v : log1pf(__expf(v));   // softplus
    } else if (mode == 2) {
      v += res[(size_t)row * ldres + col];      // residual add
    }
    C[(size_t)row * ldc + col] = v;
  }
}

#if USE_ASYNC_LDS
// one 16-byte async global->LDS copy (gfx1250 GLOBAL_LOAD_ASYNC_TO_LDS_B128)
__device__ __forceinline__ void async_copy16(const void* g, void* l) {
  __builtin_amdgcn_global_load_async_to_lds_b128(
      (__attribute__((address_space(1))) v4i*)g,
      (__attribute__((address_space(3))) v4i*)l, 0, 0);
}

template <int N>
__device__ __forceinline__ void wait_async() {
#if __has_builtin(__builtin_amdgcn_s_wait_asynccnt)
  __builtin_amdgcn_s_wait_asynccnt(N);
#else
  asm volatile("s_wait_asynccnt %0" :: "i"(N) : "memory");
#endif
}
#endif

// ---------------------------------------------------------------------------
// Generic GEMM: C[M,N] = A[M,K] * B[N,K]^T, f16 in / f32 out.
// Block = 256 threads (8 waves), tile 128(M) x 64(N); each wave owns a
// 32x32 sub-tile = 2x2 v_wmma_f32_16x16x32_f16 per K-step.
// Async-LDS path: double-buffered 128x32 A-tile + 64x32 B-tile staged with
// GLOBAL_LOAD_ASYNC_TO_LDS_B128; per-thread s_wait_asynccnt<=3 retires the
// previous stage (async loads complete in order), then a workgroup barrier
// makes the whole tile visible.
// ---------------------------------------------------------------------------
#if USE_ASYNC_LDS
__global__ void __launch_bounds__(256)
wmma_gemm_128x64(const _Float16* __restrict__ A, const _Float16* __restrict__ B,
                 float* __restrict__ C, int K, int ldc, int mode,
                 const float* __restrict__ bias,
                 const float* __restrict__ res, int ldres) {
  __shared__ __align__(16) _Float16 smA[2][128 * 32];   // 8 KB per buffer
  __shared__ __align__(16) _Float16 smB[2][64 * 32];    // 4 KB per buffer
  const int tid = threadIdx.x;
  const int wid = tid >> 5;
  const int mw  = (wid & 3) * 32;
  const int nw  = (wid >> 2) * 32;
  const int Mbase = blockIdx.y * 128;
  const int Nbase = blockIdx.x * 64;

  // 16B copy chunks: 4 chunks per 32-wide row
  const int crow = tid >> 2;          // 0..63
  const int ccol = (tid & 3) * 8;     // half-element offset

  v8f a00 = {}, a01 = {}, a10 = {}, a11 = {};
  const int nk = K >> 5;

  // stage issuer: 3 async b128 per thread (2 for A's 128 rows, 1 for B's 64)
  auto issue = [&](int kt, int buf) {
    const int k = kt * 32;
    async_copy16(A + (size_t)(Mbase + crow) * K + k + ccol,
                 &smA[buf][crow * 32 + ccol]);
    async_copy16(A + (size_t)(Mbase + crow + 64) * K + k + ccol,
                 &smA[buf][(crow + 64) * 32 + ccol]);
    async_copy16(B + (size_t)(Nbase + crow) * K + k + ccol,
                 &smB[buf][crow * 32 + ccol]);
  };

  issue(0, 0);
  for (int kt = 0; kt < nk; ++kt) {
    const int buf = kt & 1;
    if (kt + 1 < nk) { issue(kt + 1, buf ^ 1); wait_async<3>(); }
    else             { wait_async<0>(); }
    __syncthreads();                         // stage kt fully in LDS
    v16h fa0 = load_frag_lds(smA[buf], mw);
    v16h fa1 = load_frag_lds(smA[buf], mw + 16);
    v16h fb0 = load_frag_lds(smB[buf], nw);
    v16h fb1 = load_frag_lds(smB[buf], nw + 16);
    a00 = __builtin_amdgcn_wmma_f32_16x16x32_f16(false, fa0, false, fb0, (short)0, a00, false, false);
    a01 = __builtin_amdgcn_wmma_f32_16x16x32_f16(false, fa0, false, fb1, (short)0, a01, false, false);
    a10 = __builtin_amdgcn_wmma_f32_16x16x32_f16(false, fa1, false, fb0, (short)0, a10, false, false);
    a11 = __builtin_amdgcn_wmma_f32_16x16x32_f16(false, fa1, false, fb1, (short)0, a11, false, false);
    __syncthreads();                         // reads done before next overwrite
  }
  store_acc(C, ldc, mw + Mbase,      nw + Nbase,      a00, mode, bias, res, ldres);
  store_acc(C, ldc, mw + Mbase,      nw + Nbase + 16, a01, mode, bias, res, ldres);
  store_acc(C, ldc, mw + Mbase + 16, nw + Nbase,      a10, mode, bias, res, ldres);
  store_acc(C, ldc, mw + Mbase + 16, nw + Nbase + 16, a11, mode, bias, res, ldres);
}
#else
__global__ void __launch_bounds__(256)
wmma_gemm_128x64(const _Float16* __restrict__ A, const _Float16* __restrict__ B,
                 float* __restrict__ C, int K, int ldc, int mode,
                 const float* __restrict__ bias,
                 const float* __restrict__ res, int ldres) {
  const int wid = threadIdx.x >> 5;
  const int m0  = blockIdx.y * 128 + (wid & 3) * 32;
  const int n0  = blockIdx.x * 64  + (wid >> 2) * 32;
  v8f a00 = {}, a01 = {}, a10 = {}, a11 = {};
  for (int k = 0; k < K; k += 32) {
    __builtin_prefetch(A + (size_t)m0 * K + k + 128, 0, 1);
    __builtin_prefetch(B + (size_t)n0 * K + k + 128, 0, 1);
    v16h fa0 = load_frag_k(A, K, m0,      k);
    v16h fa1 = load_frag_k(A, K, m0 + 16, k);
    v16h fb0 = load_frag_k(B, K, n0,      k);
    v16h fb1 = load_frag_k(B, K, n0 + 16, k);
    a00 = __builtin_amdgcn_wmma_f32_16x16x32_f16(false, fa0, false, fb0, (short)0, a00, false, false);
    a01 = __builtin_amdgcn_wmma_f32_16x16x32_f16(false, fa0, false, fb1, (short)0, a01, false, false);
    a10 = __builtin_amdgcn_wmma_f32_16x16x32_f16(false, fa1, false, fb0, (short)0, a10, false, false);
    a11 = __builtin_amdgcn_wmma_f32_16x16x32_f16(false, fa1, false, fb1, (short)0, a11, false, false);
  }
  store_acc(C, ldc, m0,      n0,      a00, mode, bias, res, ldres);
  store_acc(C, ldc, m0,      n0 + 16, a01, mode, bias, res, ldres);
  store_acc(C, ldc, m0 + 16, n0,      a10, mode, bias, res, ldres);
  store_acc(C, ldc, m0 + 16, n0 + 16, a11, mode, bias, res, ldres);
}
#endif

// ---------------------------------------------------------------------------
// x_proj GEMM: N = 96 (6 tiles of 16). Wave owns 16 rows x 96 cols.
// Writes proj (f32) plus an f16 copy of the dt columns (0..63) for dt_proj.
// ---------------------------------------------------------------------------
__global__ void __launch_bounds__(256)
wmma_gemm_xproj(const _Float16* __restrict__ A, const _Float16* __restrict__ B,
                float* __restrict__ proj, _Float16* __restrict__ dtraw) {
  const int wid = threadIdx.x >> 5;
  const int m0  = blockIdx.x * 128 + wid * 16;
  v8f acc[6] = {};
  for (int k = 0; k < DINNER; k += 32) {
    v16h fa = load_frag_k(A, DINNER, m0, k);
#pragma unroll
    for (int j = 0; j < 6; ++j) {
      v16h fb = load_frag_k(B, DINNER, j * 16, k);
      acc[j] = __builtin_amdgcn_wmma_f32_16x16x32_f16(false, fa, false, fb, (short)0, acc[j], false, false);
    }
  }
  const int lane = threadIdx.x & 31;
  const int cl   = lane & 15;
  const int rb   = m0 + (lane >> 4) * 8;
#pragma unroll
  for (int j = 0; j < 6; ++j) {
#pragma unroll
    for (int r = 0; r < 8; ++r) {
      const int row = rb + r;
      const int col = j * 16 + cl;
      const float v = acc[j][r];
      proj[(size_t)row * NPROJ + col] = v;
      if (col < DTRANK) dtraw[(size_t)row * DTRANK + col] = (_Float16)v;
    }
  }
}

// --------------------------- elementwise kernels ---------------------------
__global__ void f32_to_f16(const float* __restrict__ in, _Float16* __restrict__ out, int n) {
  const int i = blockIdx.x * 256 + threadIdx.x;
  if (i < n) out[i] = (_Float16)in[i];
}

__global__ void __launch_bounds__(256)
layernorm_kernel(const float* __restrict__ x, const float* __restrict__ w,
                 const float* __restrict__ b, _Float16* __restrict__ xn_h) {
  __shared__ float s1[256], s2[256];
  const int l = blockIdx.x;
  const int tid = threadIdx.x;
  const float* row = x + (size_t)l * DMODEL;
  float sum = 0.f, sq = 0.f;
  for (int i = tid; i < DMODEL; i += 256) { const float v = row[i]; sum += v; sq += v * v; }
  s1[tid] = sum; s2[tid] = sq;
  __syncthreads();
  for (int s = 128; s > 0; s >>= 1) {
    if (tid < s) { s1[tid] += s1[tid + s]; s2[tid] += s2[tid + s]; }
    __syncthreads();
  }
  const float mu   = s1[0] * (1.f / DMODEL);
  const float var  = s2[0] * (1.f / DMODEL) - mu * mu;   // biased, like torch
  const float rstd = rsqrtf(var + 1e-5f);
  for (int i = tid; i < DMODEL; i += 256)
    xn_h[(size_t)l * DMODEL + i] = (_Float16)((row[i] - mu) * rstd * w[i] + b[i]);
}

__global__ void __launch_bounds__(256)
conv_silu_kernel(const float* __restrict__ xz, const float* __restrict__ cw,
                 const float* __restrict__ cb, float* __restrict__ xconv,
                 _Float16* __restrict__ xconv_h) {
  const int idx = blockIdx.x * 256 + threadIdx.x;   // SEQ*DINNER threads
  const int l = idx >> 11;          // / DINNER
  const int d = idx & (DINNER - 1);
  float acc = cb[d];
#pragma unroll
  for (int k = 0; k < 4; ++k) {
    const int t = l - 3 + k;        // causal pad of D_CONV-1
    if (t >= 0) acc += xz[(size_t)t * DIN2 + d] * cw[d * 4 + k];
  }
  const float s = acc / (1.f + __expf(-acc));       // SiLU
  xconv[idx]   = s;
  xconv_h[idx] = (_Float16)s;
}

// ---------------------------------------------------------------------------
// Selective scan: h_t = exp(dt*A)*h_{t-1} + dt*B_t*u_t ; y_t = <h_t, C_t> + D*u_t
// One 16-lane group per channel d (lane n = state index); 2 channels / wave32.
// Fuses the SiLU(z) gate and writes f16 for out_proj.
// ---------------------------------------------------------------------------
__global__ void __launch_bounds__(256)
scan_kernel(const float* __restrict__ dtv, const float* __restrict__ u,
            const float* __restrict__ proj, const float* __restrict__ xz,
            const float* __restrict__ A_log, const float* __restrict__ Dskip,
            _Float16* __restrict__ y_h) {
  const int lane = threadIdx.x & 31;
  const int wid  = threadIdx.x >> 5;
  const int grp  = lane >> 4;
  const int n    = lane & 15;
  const int d    = blockIdx.x * 16 + wid * 2 + grp;
  const float Ad = -__expf(A_log[d * DSTATE + n]);
  const float Dd = Dskip[d];
  float h = 0.f;
  for (int t = 0; t < SEQ; ++t) {
    const float dt = dtv[(size_t)t * DINNER + d];
    const float uv = u  [(size_t)t * DINNER + d];
    const float Bn = proj[(size_t)t * NPROJ + DTRANK + n];
    const float Cn = proj[(size_t)t * NPROJ + DTRANK + DSTATE + n];
    h = __expf(dt * Ad) * h + dt * Bn * uv;
    float y = h * Cn;                       // reduce over n (stays in 16-lane group)
    y += __shfl_xor(y, 1, 32);
    y += __shfl_xor(y, 2, 32);
    y += __shfl_xor(y, 4, 32);
    y += __shfl_xor(y, 8, 32);
    if (n == 0) {
      y += uv * Dd;
      const float zv = xz[(size_t)t * DIN2 + DINNER + d];
      const float g  = zv / (1.f + __expf(-zv));      // SiLU(z)
      y_h[(size_t)t * DINNER + d] = (_Float16)(y * g);
    }
  }
}

// ---------------------------------------------------------------------------
extern "C" void kernel_launch(void* const* d_in, const int* in_sizes, int n_in,
                              void* d_out, int out_size, void* d_ws, size_t ws_size,
                              hipStream_t stream) {
  (void)in_sizes; (void)n_in; (void)out_size; (void)ws_size;
  const float* x      = (const float*)d_in[0];
  const float* ln_w   = (const float*)d_in[1];
  const float* ln_b   = (const float*)d_in[2];
  const float* W_in   = (const float*)d_in[3];
  const float* conv_w = (const float*)d_in[4];
  const float* conv_b = (const float*)d_in[5];
  const float* W_xp   = (const float*)d_in[6];
  const float* W_dt   = (const float*)d_in[7];
  const float* b_dt   = (const float*)d_in[8];
  const float* A_log  = (const float*)d_in[9];
  const float* Dskip  = (const float*)d_in[10];
  const float* W_out  = (const float*)d_in[11];
  float* out = (float*)d_out;

  char* ws = (char*)d_ws;
  size_t off = 0;
  auto take = [&](size_t bytes) -> char* {
    char* p = ws + off;
    off += (bytes + 255) & ~(size_t)255;
    return p;
  };
  _Float16* W_in_h  = (_Float16*)take((size_t)DIN2   * DMODEL * 2);
  _Float16* W_xp_h  = (_Float16*)take((size_t)NPROJ  * DINNER * 2);
  _Float16* W_dt_h  = (_Float16*)take((size_t)DINNER * DTRANK * 2);
  _Float16* W_out_h = (_Float16*)take((size_t)DMODEL * DINNER * 2);
  _Float16* xn_h    = (_Float16*)take((size_t)SEQ * DMODEL * 2);
  float*    xz      = (float*)   take((size_t)SEQ * DIN2   * 4);
  float*    xconv   = (float*)   take((size_t)SEQ * DINNER * 4);
  _Float16* xconv_h = (_Float16*)take((size_t)SEQ * DINNER * 2);
  float*    proj    = (float*)   take((size_t)SEQ * NPROJ  * 4);
  _Float16* dtraw_h = (_Float16*)take((size_t)SEQ * DTRANK * 2);
  float*    dtv     = (float*)   take((size_t)SEQ * DINNER * 4);
  _Float16* y_h     = (_Float16*)take((size_t)SEQ * DINNER * 2);

  const int T = 256;
  // weight f32 -> f16 (re-done every call; deterministic, no cached state)
  f32_to_f16<<<(DIN2 * DMODEL  + T - 1) / T, T, 0, stream>>>(W_in,  W_in_h,  DIN2 * DMODEL);
  f32_to_f16<<<(NPROJ * DINNER + T - 1) / T, T, 0, stream>>>(W_xp,  W_xp_h,  NPROJ * DINNER);
  f32_to_f16<<<(DINNER * DTRANK+ T - 1) / T, T, 0, stream>>>(W_dt,  W_dt_h,  DINNER * DTRANK);
  f32_to_f16<<<(DMODEL * DINNER+ T - 1) / T, T, 0, stream>>>(W_out, W_out_h, DMODEL * DINNER);

  // 1. pre-norm
  layernorm_kernel<<<SEQ, T, 0, stream>>>(x, ln_w, ln_b, xn_h);
  // 2. in_proj: xz[2048,4096] = xn * W_in^T
  wmma_gemm_128x64<<<dim3(DIN2 / 64, SEQ / 128), T, 0, stream>>>(
      xn_h, W_in_h, xz, DMODEL, DIN2, 0, nullptr, nullptr, 0);
  // 3. causal depthwise conv + SiLU
  conv_silu_kernel<<<(SEQ * DINNER) / T, T, 0, stream>>>(xz, conv_w, conv_b, xconv, xconv_h);
  // 4. x_proj: proj[2048,96] = xconv * W_xproj^T (also f16 dt columns)
  wmma_gemm_xproj<<<SEQ / 128, T, 0, stream>>>(xconv_h, W_xp_h, proj, dtraw_h);
  // 5. dt_proj + softplus: dtv[2048,2048]
  wmma_gemm_128x64<<<dim3(DINNER / 64, SEQ / 128), T, 0, stream>>>(
      dtraw_h, W_dt_h, dtv, DTRANK, DINNER, 1, b_dt, nullptr, 0);
  // 6. selective scan + SiLU(z) gate -> y_h (f16)
  scan_kernel<<<DINNER / 16, T, 0, stream>>>(dtv, xconv, proj, xz, A_log, Dskip, y_h);
  // 7. out_proj + residual -> out
  wmma_gemm_128x64<<<dim3(DMODEL / 64, SEQ / 128), T, 0, stream>>>(
      y_h, W_out_h, out, DINNER, DMODEL, 2, nullptr, x, DMODEL);
}
